// MaskedTransformerEncoderLayer_15814069584505
// MI455X (gfx1250) — matrix operand order repaired and checked by reading, not compiled, not run on hardware
//
#include <hip/hip_runtime.h>
#include <hip/hip_bf16.h>
#include <math.h>

// ---------------------------------------------------------------------------
// MI455X / gfx1250, wave32.
//   - GEMMs: v_wmma_f32_16x16x32_bf16
//   - GEMM activation tiles: TENSOR_LOAD_TO_LDS (TDM, TENSORcnt)
//   - Attention K/V tiles:   GLOBAL_LOAD_ASYNC_TO_LDS_B128 (ASYNCcnt)
// ---------------------------------------------------------------------------

typedef __bf16 bf16_t;
typedef bf16_t v16bf __attribute__((ext_vector_type(16)));
typedef bf16_t v8bf  __attribute__((ext_vector_type(8)));
typedef float  v8f   __attribute__((ext_vector_type(8)));
typedef int    v4i_  __attribute__((ext_vector_type(4)));
typedef int    v8i_  __attribute__((ext_vector_type(8)));
typedef unsigned int v4u_ __attribute__((ext_vector_type(4)));

#define B_  8
#define N_  1024
#define D_  1024
#define H_  16
#define HD_ 64
#define F_  4096
#define T_  (B_ * N_)

// ---- CDNA5 async global->LDS copy (16B) -----------------------------------
#if __has_builtin(__builtin_amdgcn_global_load_async_to_lds_b128)
#define HAS_ASYNC 1
#define ASYNC_CP16(g, l)                                                      \
  __builtin_amdgcn_global_load_async_to_lds_b128(                             \
      (__attribute__((address_space(1))) v4i_*)(uintptr_t)(g),                \
      (__attribute__((address_space(3))) v4i_*)(unsigned)(uintptr_t)(l), 0, 0)
#else
#define HAS_ASYNC 0
#warning "gfx1250 async-to-LDS builtin not found; using synchronous LDS staging"
#define ASYNC_CP16(g, l) (*(v8bf*)(l) = *(const v8bf*)(g))
#endif

#if HAS_ASYNC
#if __has_builtin(__builtin_amdgcn_s_wait_asynccnt)
#define WAIT_ASYNC(n) __builtin_amdgcn_s_wait_asynccnt(n)
#else
#define WAIT_ASYNC(n) asm volatile("s_wait_asynccnt %0" ::"i"(n) : "memory")
#endif
#else
#define WAIT_ASYNC(n) ((void)0)
#endif

// ---- CDNA5 Tensor Data Mover (TDM) ----------------------------------------
// This toolchain exposes the 6-arg form:
//   (uint32x4 g0, int32x8 g1, int32x4 g2, int32x4 g3, int32x8, i32 cpol)
#if __has_builtin(__builtin_amdgcn_tensor_load_to_lds) && \
    __has_builtin(__builtin_amdgcn_s_wait_tensorcnt)
#define HAS_TDM 1
#else
#define HAS_TDM 0
#warning "gfx1250 TDM builtin not found; GEMM staging falls back to async path"
#endif

static __device__ __forceinline__ v8f vz8() {
  v8f z;
#pragma unroll
  for (int i = 0; i < 8; ++i) z[i] = 0.f;
  return z;
}

static __device__ __forceinline__ v8bf cvt8(v8f a) {
  v8bf r;
#pragma unroll
  for (int i = 0; i < 8; ++i) r[i] = (bf16_t)a[i];
  return r;
}

static __device__ __forceinline__ v8f wmma_bf16(v16bf a, v16bf b, v8f c) {
  // D = A(16x32 bf16) x B(32x16 bf16) + C(16x16 f32)
  return __builtin_amdgcn_wmma_f32_16x16x32_bf16(false, a, false, b,
                                                 (short)0, c, false, false);
}

// A-fragment (16x32, 16-bit): lane(row=lane&15, h=lane>>4):
//   vector elems 0..15 = K {16h+0..15}
static __device__ __forceinline__ v16bf load_a_frag(const bf16_t* row, int kb,
                                                    int lh) {
  v8bf lo = *(const v8bf*)(row + kb + 16 * lh);
  v8bf hi = *(const v8bf*)(row + kb + 16 * lh + 8);
  v16bf a;
#pragma unroll
  for (int i = 0; i < 8; ++i) { a[i] = lo[i]; a[8 + i] = hi[i]; }
  return a;
}

// Per-thread async staging of a 64x64-elem bf16 tile (8KB): 2 b128/thread.
static __device__ __forceinline__ void stage_tile64(bf16_t* dst,
                                                    const bf16_t* src, int ldx,
                                                    int tid) {
  int byteoff = tid * 32;            // 256 threads * 32B = 8KB
  int row = byteoff >> 7;            // 128B per row
  int col = byteoff & 127;
  const char* g = (const char*)src + (size_t)row * (size_t)(ldx * 2) + col;
  char* l = (char*)dst + byteoff;
  ASYNC_CP16(g, l);
  ASYNC_CP16(g + 16, l + 16);
}

#if HAS_TDM
// One TDM descriptor moves the whole 64x64x2B tile (tile start at srcGlobal,
// row stride ldx elems) into LDS. D# packing per CDNA5 ISA ch. 8.3/8.4:
//   group0: count=1 | lds_addr | global_addr[56:0] | type=2
//   group1: data_size=1(2B), tensor_dim0=ldx, tensor_dim1=T_,
//           tile_dim0=64, tile_dim1=64, tensor_dim0_stride=ldx
// workgroup_mask=0 (not in a cluster); groups 2/3 + trailing group zero (2D).
static __device__ __forceinline__ void tdm_load_tile64(bf16_t* dstLds,
                                                       const bf16_t* srcGlobal,
                                                       int ldx) {
  unsigned lds_addr = (unsigned)(uintptr_t)dstLds;  // flat[31:0] == LDS offset
  unsigned long long ga = (unsigned long long)(uintptr_t)srcGlobal;
  v4u_ g0;
  g0[0] = 1u;                                      // count=1
  g0[1] = lds_addr;
  g0[2] = (unsigned)(ga & 0xffffffffu);            // global_addr[31:0]
  g0[3] = (unsigned)((ga >> 32) & 0x01ffffffu) | (2u << 30);  // [56:32]|type=2
  unsigned dim0 = (unsigned)ldx;
  v8i_ g1;
  g1[0] = (int)(1u << 16);                         // data_size=1 (2 bytes)
  g1[1] = (int)((dim0 & 0xffffu) << 16);           // tensor_dim0[15:0]
  g1[2] = (int)((dim0 >> 16) & 0xffffu);           // tensor_dim0[31:16]
  g1[2] |= (int)(((unsigned)T_ & 0xffffu) << 16);  // tensor_dim1[15:0]
  g1[3] = (int)(((unsigned)T_ >> 16) & 0xffffu)    // tensor_dim1[31:16]
          | (int)(64u << 16);                      // tile_dim0 = 64
  g1[4] = (int)64u;                                // tile_dim1=64, tile_dim2=0
  g1[5] = (int)dim0;                               // tensor_dim0_stride[31:0]
  g1[6] = 0;                                       // stride0[47:32], stride1[15:0]
  g1[7] = 0;                                       // stride1[47:16]
  v4i_ gz4 = {0, 0, 0, 0};
  v8i_ gz8 = {0, 0, 0, 0, 0, 0, 0, 0};
  __builtin_amdgcn_tensor_load_to_lds(g0, g1, gz4, gz4, gz8, 0);
}
#define STAGE_X(dst, src, ldx, tid) \
  do { if ((tid) < 32) tdm_load_tile64((dst), (src), (ldx)); } while (0)
#define WAIT_X() __builtin_amdgcn_s_wait_tensorcnt(1)
#else
#define STAGE_X(dst, src, ldx, tid) stage_tile64((dst), (src), (ldx), (tid))
#define WAIT_X() WAIT_ASYNC(2)
#endif

// ---------------------------------------------------------------------------
// Block-cooperative C^T = W * X^T core.
// All 8 waves share one 64-token X tile, double-buffered in LDS via TDM;
// per-wave weight A-fragments pipelined from global (issued pre-barrier).
// Wave computes 16 channels x 64 tokens (4 accum tiles).
// ---------------------------------------------------------------------------
template <int K, int LDW, int LDX>
static __device__ __forceinline__ void gemm_core_lds(
    const bf16_t* __restrict__ W, const bf16_t* __restrict__ X, bf16_t* xtile,
    int ch0, int tok0, int lr, int lh, int tid, v8f acc[4]) {
  const bf16_t* arow = W + (size_t)(ch0 + lr) * LDW;
  const bf16_t* xbase = X + (size_t)tok0 * LDX;
  STAGE_X(xtile, xbase, LDX, tid);
  for (int kb = 0; kb < K; kb += 64) {
    bf16_t* cur = xtile + ((kb >> 6) & 1) * (64 * 64);
    if (kb + 64 < K)
      STAGE_X(xtile + (((kb >> 6) + 1) & 1) * (64 * 64), xbase + kb + 64, LDX,
              tid);
    // Weight fragments issued before the barrier: latency hides under sync.
    v16bf a0 = load_a_frag(arow, kb, lh);
    v16bf a1 = load_a_frag(arow, kb + 32, lh);
    WAIT_X();            // next-buffer DMA may stay in flight (in-order)
    __syncthreads();
    const bf16_t* brow = cur + (size_t)lr * 64 + 16 * lh;
    v16bf bf[4];
#pragma unroll
    for (int t = 0; t < 4; ++t) bf[t] = *(const v16bf*)(brow + t * 16 * 64);
#pragma unroll
    for (int t = 0; t < 4; ++t) acc[t] = wmma_bf16(a0, bf[t], acc[t]);
#pragma unroll
    for (int t = 0; t < 4; ++t) bf[t] = *(const v16bf*)(brow + t * 16 * 64 + 32);
#pragma unroll
    for (int t = 0; t < 4; ++t) acc[t] = wmma_bf16(a1, bf[t], acc[t]);
    __syncthreads();     // protect buffer being restaged next iteration
  }
}

// ---------------------------------------------------------------------------
// Elementwise fp32 -> bf16 weight conversion
// ---------------------------------------------------------------------------
__global__ void cvt_bf16_kernel(const float* __restrict__ in,
                                bf16_t* __restrict__ out, int n) {
  for (int i = blockIdx.x * blockDim.x + threadIdx.x; i < n;
       i += gridDim.x * blockDim.x)
    out[i] = (bf16_t)in[i];
}

// ---------------------------------------------------------------------------
// LayerNorm: one 256-thread block per token row; optional fp32 copy (the
// final residual uses the *normed* activation).
// ---------------------------------------------------------------------------
__global__ __launch_bounds__(256) void ln_kernel(const float* __restrict__ x,
                                                 const float* __restrict__ g,
                                                 const float* __restrict__ beta,
                                                 bf16_t* __restrict__ obf,
                                                 float* of32, int Dd) {
  int row = blockIdx.x;
  const float* xr = x + (size_t)row * Dd;
  float s = 0.f, s2 = 0.f;
  for (int i = threadIdx.x; i < Dd; i += 256) {
    float v = xr[i];
    s += v; s2 += v * v;
  }
#pragma unroll
  for (int o = 16; o >= 1; o >>= 1) {
    s  += __shfl_xor(s,  o, 32);
    s2 += __shfl_xor(s2, o, 32);
  }
  __shared__ float red[2][8];
  if ((threadIdx.x & 31) == 0) {
    red[0][threadIdx.x >> 5] = s;
    red[1][threadIdx.x >> 5] = s2;
  }
  __syncthreads();
  float ts = 0.f, ts2 = 0.f;
#pragma unroll
  for (int i = 0; i < 8; ++i) { ts += red[0][i]; ts2 += red[1][i]; }
  float mean = ts / Dd;
  float var  = ts2 / Dd - mean * mean;
  float inv  = rsqrtf(var + 1e-5f);
  for (int i = threadIdx.x; i < Dd; i += 256) {
    float v = (xr[i] - mean) * inv * g[i] + beta[i];
    obf[(size_t)row * Dd + i] = (bf16_t)v;
    if (of32) of32[(size_t)row * Dd + i] = v;
  }
}

// ---------------------------------------------------------------------------
// QKV GEMM: qkv^T = Wqkv * h^T. Q,K stored [b][h][n][d]; V stored transposed
// [b][h][d][n] so each half of the attention PV A-fragment is a contiguous 16B load.
// ---------------------------------------------------------------------------
__global__ __launch_bounds__(256) void qkv_kernel(const bf16_t* __restrict__ Wq,
                                                  const bf16_t* __restrict__ h,
                                                  bf16_t* __restrict__ q,
                                                  bf16_t* __restrict__ kk,
                                                  bf16_t* __restrict__ vt) {
  __shared__ bf16_t xtile[2][64 * 64];
  int tid = threadIdx.x;
  int lane = tid & 31, lr = lane & 15, lh = lane >> 4, widx = tid >> 5;
  int cb = blockIdx.x % 24, tt = blockIdx.x / 24;  // 192 ch-tiles / 8 waves
  int ch0 = (cb * 8 + widx) * 16, tok0 = tt * 64;
  v8f acc[4] = {vz8(), vz8(), vz8(), vz8()};
  gemm_core_lds<D_, D_, D_>(Wq, h, &xtile[0][0], ch0, tok0, lr, lh, tid, acc);

  int which = ch0 >> 10, rem = ch0 & 1023, head = rem >> 6, dd0 = rem & 63;
#pragma unroll
  for (int t = 0; t < 4; ++t) {
    int tok = tok0 + t * 16 + lr;
    int bb = tok >> 10, n = tok & 1023;
    if (which < 2) {
      bf16_t* dst = (which == 0 ? q : kk) +
                    ((size_t)(bb * H_ + head) * N_ + n) * HD_ + dd0 + 8 * lh;
      *(v8bf*)dst = cvt8(acc[t]);
    } else {
#pragma unroll
      for (int r = 0; r < 8; ++r)
        vt[((size_t)(bb * H_ + head) * HD_ + dd0 + 8 * lh + r) * N_ + n] =
            (bf16_t)acc[t][r];
    }
  }
}

// ---------------------------------------------------------------------------
// Flash attention: one wave per 16-query tile, 8 waves/block share (b,head).
// S^T = K*Q^T so each lane owns one query column; exp(S^T) accumulators map
// register-for-register onto the B-fragment of O^T = V^T*P^T. K and V^T
// chunks are async-staged (ASYNCcnt path) into LDS once per block.
// ---------------------------------------------------------------------------
static __device__ __forceinline__ void stage_kv(bf16_t* kdst, bf16_t* vdst,
                                                const bf16_t* kbp,
                                                const bf16_t* vbp, int kc,
                                                int tid) {
  {  // K tile: 32 keys x 64 dims (128B rows) = 4KB
    int byteoff = tid * 16;
    int row = byteoff >> 7, col = byteoff & 127;
    const char* g = (const char*)(kbp + (size_t)(kc + row) * HD_) + col;
    ASYNC_CP16(g, (char*)kdst + byteoff);
  }
  {  // V^T tile: 64 dims x 32 keys (64B rows) = 4KB
    int byteoff = tid * 16;
    int row = byteoff >> 6, col = byteoff & 63;
    const char* g = (const char*)(vbp + (size_t)row * N_ + kc) + col;
    ASYNC_CP16(g, (char*)vdst + byteoff);
  }
}

__global__ __launch_bounds__(256) void attn_kernel(
    const bf16_t* __restrict__ Q, const bf16_t* __restrict__ Km,
    const bf16_t* __restrict__ Vt, const unsigned char* __restrict__ mask,
    bf16_t* __restrict__ aout) {
  __shared__ bf16_t ktile[2][32 * 64];
  __shared__ bf16_t vtile[2][64 * 32];
  __shared__ unsigned char smask[N_];
  int tid = threadIdx.x;
  int lane = tid & 31, lr = lane & 15, lh = lane >> 4, widx = tid >> 5;
  int wid = blockIdx.x * 8 + widx;
  int qt = wid & 63;       // 64 query tiles per (b,h)
  int bh = wid >> 6;       // uniform across the block (64 % 8 == 0)
  int b = bh >> 4, head = bh & 15;

  for (int i = tid; i < N_; i += 256) smask[i] = mask[(size_t)b * N_ + i];

  const bf16_t* qb = Q  + (size_t)bh * N_ * HD_;
  const bf16_t* kbp = Km + (size_t)bh * N_ * HD_;
  const bf16_t* vbp = Vt + (size_t)bh * HD_ * N_;
  int q0 = qt * 16;

  // Q^T B-fragments (reused over every key chunk)
  v16bf qf0 = *(const v16bf*)(qb + (size_t)(q0 + lr) * HD_ + 16 * lh);
  v16bf qf1 = *(const v16bf*)(qb + (size_t)(q0 + lr) * HD_ + 32 + 16 * lh);

  stage_kv(ktile[0], vtile[0], kbp, vbp, 0, tid);
  __syncthreads();  // publish smask (and order prologue staging)
  bool qkeep = smask[q0 + lr] != 0;

  const float NEG = -3.402823466e38f;
  const float scale = 0.125f;  // 1/sqrt(64)
  float mrow = NEG, lrow = 0.f;
  v8f o0 = vz8(), o1 = vz8(), o2 = vz8(), o3 = vz8();

  for (int kc = 0; kc < N_; kc += 32) {
    int cur = (kc >> 5) & 1;
    if (kc + 32 < N_)
      stage_kv(ktile[cur ^ 1], vtile[cur ^ 1], kbp, vbp, kc + 32, tid);
    WAIT_ASYNC(2);
    __syncthreads();
    const bf16_t* kl = ktile[cur];
    const bf16_t* vl = vtile[cur];

    // S^T tiles from LDS: tile0 = keys kc..kc+15, tile1 = keys kc+16..kc+31
    v8f s0 = vz8(), s1 = vz8();
    s0 = wmma_bf16(load_a_frag(kl + (size_t)lr * 64, 0, lh), qf0, s0);
    s0 = wmma_bf16(load_a_frag(kl + (size_t)lr * 64, 32, lh), qf1, s0);
    s1 = wmma_bf16(load_a_frag(kl + (size_t)(16 + lr) * 64, 0, lh), qf0, s1);
    s1 = wmma_bf16(load_a_frag(kl + (size_t)(16 + lr) * 64, 32, lh), qf1, s1);

    float cmax = NEG;
#pragma unroll
    for (int r = 0; r < 8; ++r) {
      int key = kc + 8 * lh + r;
      float v0 = (qkeep && smask[key])      ? s0[r] * scale : NEG;
      float v1 = (qkeep && smask[key + 16]) ? s1[r] * scale : NEG;
      s0[r] = v0; s1[r] = v1;
      cmax = fmaxf(cmax, fmaxf(v0, v1));
    }
    cmax = fmaxf(cmax, __shfl_xor(cmax, 16, 32));
    float mnew = fmaxf(mrow, cmax);
    float corr = __expf(mrow - mnew);
    float csum = 0.f;
#pragma unroll
    for (int r = 0; r < 8; ++r) {
      float p0 = __expf(s0[r] - mnew);
      float p1 = __expf(s1[r] - mnew);
      s0[r] = p0; s1[r] = p1;
      csum += p0 + p1;
    }
    csum += __shfl_xor(csum, 16, 32);
    lrow = lrow * corr + csum;
    mrow = mnew;
#pragma unroll
    for (int r = 0; r < 8; ++r) { o0[r] *= corr; o1[r] *= corr; o2[r] *= corr; o3[r] *= corr; }

    // P^T B-fragment straight from the S^T accumulators (no cross-lane moves)
    v16bf pf;
#pragma unroll
    for (int i = 0; i < 8; ++i) { pf[i] = (bf16_t)s0[i]; pf[8 + i] = (bf16_t)s1[i]; }

    // O^T += V^T * P^T (4 dim tiles of 16), V^T from LDS
    o0 = wmma_bf16(__builtin_shufflevector(*(const v8bf*)(vl + (size_t)( 0 + lr) * 32 + 8 * lh), *(const v8bf*)(vl + (size_t)( 0 + lr) * 32 + 16 + 8 * lh), 0, 1, 2, 3, 4, 5, 6, 7, 8, 9, 10, 11, 12, 13, 14, 15), pf, o0);
    o1 = wmma_bf16(__builtin_shufflevector(*(const v8bf*)(vl + (size_t)(16 + lr) * 32 + 8 * lh), *(const v8bf*)(vl + (size_t)(16 + lr) * 32 + 16 + 8 * lh), 0, 1, 2, 3, 4, 5, 6, 7, 8, 9, 10, 11, 12, 13, 14, 15), pf, o1);
    o2 = wmma_bf16(__builtin_shufflevector(*(const v8bf*)(vl + (size_t)(32 + lr) * 32 + 8 * lh), *(const v8bf*)(vl + (size_t)(32 + lr) * 32 + 16 + 8 * lh), 0, 1, 2, 3, 4, 5, 6, 7, 8, 9, 10, 11, 12, 13, 14, 15), pf, o2);
    o3 = wmma_bf16(__builtin_shufflevector(*(const v8bf*)(vl + (size_t)(48 + lr) * 32 + 8 * lh), *(const v8bf*)(vl + (size_t)(48 + lr) * 32 + 16 + 8 * lh), 0, 1, 2, 3, 4, 5, 6, 7, 8, 9, 10, 11, 12, 13, 14, 15), pf, o3);
    __syncthreads();
  }

  float invl = 1.f / lrow;
  size_t base = ((size_t)(b * N_ + q0 + lr)) * D_ + (size_t)head * HD_ + 8 * lh;
  v8f oo[4] = {o0, o1, o2, o3};
#pragma unroll
  for (int dt = 0; dt < 4; ++dt) {
    v8bf ob;
#pragma unroll
    for (int r = 0; r < 8; ++r) ob[r] = (bf16_t)(oo[dt][r] * invl);
    *(v8bf*)(aout + base + dt * 16) = ob;
  }
}

// ---------------------------------------------------------------------------
// proj: x = src + attn_out @ Wproj^T + bproj   (fp32 out)
// ---------------------------------------------------------------------------
__global__ __launch_bounds__(256) void proj_kernel(const bf16_t* __restrict__ Wp,
                                                   const bf16_t* __restrict__ A,
                                                   const float* __restrict__ src,
                                                   const float* __restrict__ bias,
                                                   float* __restrict__ xout) {
  __shared__ bf16_t xtile[2][64 * 64];
  int tid = threadIdx.x;
  int lane = tid & 31, lr = lane & 15, lh = lane >> 4, widx = tid >> 5;
  int cb = blockIdx.x % 8, tt = blockIdx.x / 8;    // 64 ch-tiles / 8 waves
  int ch0 = (cb * 8 + widx) * 16, tok0 = tt * 64;
  v8f acc[4] = {vz8(), vz8(), vz8(), vz8()};
  gemm_core_lds<D_, D_, D_>(Wp, A, &xtile[0][0], ch0, tok0, lr, lh, tid, acc);
  int chb = ch0 + 8 * lh;
#pragma unroll
  for (int t = 0; t < 4; ++t) {
    int tok = tok0 + t * 16 + lr;
    const float* s = src + (size_t)tok * D_ + chb;
    float* d = xout + (size_t)tok * D_ + chb;
#pragma unroll
    for (int r = 0; r < 8; ++r) d[r] = s[r] + acc[t][r] + bias[chb + r];
  }
}

// ---------------------------------------------------------------------------
// FFN1: y = gelu(xln @ W1^T + b1) -> bf16 (exact erf GELU)
// ---------------------------------------------------------------------------
__global__ __launch_bounds__(256) void ffn1_kernel(const bf16_t* __restrict__ W1b,
                                                   const bf16_t* __restrict__ X,
                                                   const float* __restrict__ b1,
                                                   bf16_t* __restrict__ y) {
  __shared__ bf16_t xtile[2][64 * 64];
  int tid = threadIdx.x;
  int lane = tid & 31, lr = lane & 15, lh = lane >> 4, widx = tid >> 5;
  int cb = blockIdx.x % 32, tt = blockIdx.x / 32;  // 256 ch-tiles / 8 waves
  int ch0 = (cb * 8 + widx) * 16, tok0 = tt * 64;
  v8f acc[4] = {vz8(), vz8(), vz8(), vz8()};
  gemm_core_lds<D_, D_, D_>(W1b, X, &xtile[0][0], ch0, tok0, lr, lh, tid, acc);
  int chb = ch0 + 8 * lh;
#pragma unroll
  for (int t = 0; t < 4; ++t) {
    int tok = tok0 + t * 16 + lr;
    v8bf ob;
#pragma unroll
    for (int r = 0; r < 8; ++r) {
      float v = acc[t][r] + b1[chb + r];
      v = 0.5f * v * (1.f + erff(v * 0.70710678118654752440f));
      ob[r] = (bf16_t)v;
    }
    *(v8bf*)(y + (size_t)tok * F_ + chb) = ob;
  }
}

// ---------------------------------------------------------------------------
// FFN2: out = xn + y @ W2^T + b2 (fp32 out; residual is the NORMED x)
// ---------------------------------------------------------------------------
__global__ __launch_bounds__(256) void ffn2_kernel(const bf16_t* __restrict__ W2b,
                                                   const bf16_t* __restrict__ Y,
                                                   const float* __restrict__ xn,
                                                   const float* __restrict__ b2,
                                                   float* __restrict__ out) {
  __shared__ bf16_t xtile[2][64 * 64];
  int tid = threadIdx.x;
  int lane = tid & 31, lr = lane & 15, lh = lane >> 4, widx = tid >> 5;
  int cb = blockIdx.x % 8, tt = blockIdx.x / 8;
  int ch0 = (cb * 8 + widx) * 16, tok0 = tt * 64;
  v8f acc[4] = {vz8(), vz8(), vz8(), vz8()};
  gemm_core_lds<F_, F_, F_>(W2b, Y, &xtile[0][0], ch0, tok0, lr, lh, tid, acc);
  int chb = ch0 + 8 * lh;
#pragma unroll
  for (int t = 0; t < 4; ++t) {
    int tok = tok0 + t * 16 + lr;
    const float* s = xn + (size_t)tok * D_ + chb;
    float* d = out + (size_t)tok * D_ + chb;
#pragma unroll
    for (int r = 0; r < 8; ++r) d[r] = s[r] + acc[t][r] + b2[chb + r];
  }
}

// ---------------------------------------------------------------------------
// Orchestration
// ---------------------------------------------------------------------------
extern "C" void kernel_launch(void* const* d_in, const int* in_sizes, int n_in,
                              void* d_out, int out_size, void* d_ws, size_t ws_size,
                              hipStream_t stream) {
  (void)in_sizes; (void)n_in; (void)out_size; (void)ws_size;
  const float* src          = (const float*)d_in[0];
  const unsigned char* mask = (const unsigned char*)d_in[1];  // jnp bool = 1B
  const float* Wqkv         = (const float*)d_in[2];
  const float* Wproj        = (const float*)d_in[3];
  const float* bproj        = (const float*)d_in[4];
  const float* W1           = (const float*)d_in[5];
  const float* b1           = (const float*)d_in[6];
  const float* W2           = (const float*)d_in[7];
  const float* b2           = (const float*)d_in[8];
  const float* g0           = (const float*)d_in[9];
  const float* beta0        = (const float*)d_in[10];
  const float* g1           = (const float*)d_in[11];
  const float* beta1        = (const float*)d_in[12];
  float* out = (float*)d_out;

  char* w = (char*)d_ws;
  const size_t MB = 1ull << 20;
  bf16_t* hbuf   = (bf16_t*)(w + 0);         // 16MB: h (LN0), later xln (LN1)
  bf16_t* qbuf   = (bf16_t*)(w + 16 * MB);   // 16MB
  bf16_t* kbuf   = (bf16_t*)(w + 32 * MB);   // 16MB
  bf16_t* vtbuf  = (bf16_t*)(w + 48 * MB);   // 16MB (V transposed)
  bf16_t* abuf   = (bf16_t*)(w + 64 * MB);   // 16MB attention output
  bf16_t* ybuf   = (bf16_t*)(w + 16 * MB);   // 64MB, reuses q/k/vt/abuf
  float*  xbuf   = (float*) (w + 80 * MB);   // 32MB fp32 residual / normed x
  bf16_t* wqkvb  = (bf16_t*)(w + 112 * MB);  // 6MB
  bf16_t* wprojb = (bf16_t*)(w + 118 * MB);  // 2MB
  bf16_t* w1b    = (bf16_t*)(w + 126 * MB);  // 8MB
  bf16_t* w2b    = (bf16_t*)(w + 134 * MB);  // 8MB  (peak ~142MB)

  dim3 blk(256);

  cvt_bf16_kernel<<<dim3(2048), blk, 0, stream>>>(Wqkv,  wqkvb,  3 * D_ * D_);
  cvt_bf16_kernel<<<dim3(1024), blk, 0, stream>>>(Wproj, wprojb, D_ * D_);
  cvt_bf16_kernel<<<dim3(2048), blk, 0, stream>>>(W1,    w1b,    F_ * D_);
  cvt_bf16_kernel<<<dim3(2048), blk, 0, stream>>>(W2,    w2b,    D_ * F_);

  ln_kernel<<<dim3(T_), blk, 0, stream>>>(src, g0, beta0, hbuf, (float*)nullptr, D_);

  // QKV: (192/8) channel-blocks x 128 token tiles = 3072 blocks
  qkv_kernel<<<dim3(3072), blk, 0, stream>>>(wqkvb, hbuf, qbuf, kbuf, vtbuf);

  // Attention: 8192 waves -> 1024 blocks (8 query tiles per (b,h) block)
  attn_kernel<<<dim3(1024), blk, 0, stream>>>(qbuf, kbuf, vtbuf, mask, abuf);

  // proj + residual: (64/8) x 128 = 1024 blocks
  proj_kernel<<<dim3(1024), blk, 0, stream>>>(wprojb, abuf, src, bproj, xbuf);

  // LN1: xn (fp32, in-place) + xln (bf16)
  ln_kernel<<<dim3(T_), blk, 0, stream>>>(xbuf, g1, beta1, hbuf, xbuf, D_);

  // FFN1 (GELU): (256/8) x 128 = 4096 blocks
  ffn1_kernel<<<dim3(4096), blk, 0, stream>>>(w1b, hbuf, b1, ybuf);

  // FFN2 + residual: (64/8) x 128 = 1024 blocks
  ffn2_kernel<<<dim3(1024), blk, 0, stream>>>(w2b, ybuf, xbuf, b2, out);
}
